// SelfAttention_59072980189405
// MI455X (gfx1250) — matrix operand order repaired and checked by reading, not compiled, not run on hardware
//
#include <hip/hip_runtime.h>
#include <hip/hip_bf16.h>

// Problem constants
#define S_LEN 1024
#define BATCH 8
#define HID   1024
#define NHEAD 16
#define HDIM  64

typedef __attribute__((ext_vector_type(16))) __bf16 v16bf;
typedef __attribute__((ext_vector_type(8)))  __bf16 v8bf;
typedef __attribute__((ext_vector_type(8)))  float  v8f;
typedef __attribute__((ext_vector_type(4)))  float  v4f;
typedef __attribute__((ext_vector_type(4)))  unsigned int v4u;
typedef __attribute__((ext_vector_type(8)))  int    v8i;
typedef __attribute__((ext_vector_type(4)))  int    v4i;

union FragBF {
    v16bf v;
    v8bf  h[2];
};

__device__ inline v8f wmma_bf16(const FragBF& a, const FragBF& b, v8f c) {
    return __builtin_amdgcn_wmma_f32_16x16x32_bf16(
        false, a.v, false, b.v, (short)0, c, false, false);
}

// ---------------------------------------------------------------------------
// Kernel 0: one-shot fp32 -> bf16 conversion (8 elements / thread).
// Reads 108 MB, writes 54 MB total across 6 launches: ~7 us at 23.3 TB/s.
// ---------------------------------------------------------------------------
__global__ __launch_bounds__(256)
void cvt_kernel(const float* __restrict__ src, __bf16* __restrict__ dst, int n8)
{
    int i = blockIdx.x * 256 + threadIdx.x;
    if (i >= n8) return;
    const float* p = src + (size_t)i * 8;
    v4f x0 = *(const v4f*)(p);
    v4f x1 = *(const v4f*)(p + 4);
    v8bf r;
    r[0] = (__bf16)x0[0]; r[1] = (__bf16)x0[1];
    r[2] = (__bf16)x0[2]; r[3] = (__bf16)x0[3];
    r[4] = (__bf16)x1[0]; r[5] = (__bf16)x1[1];
    r[6] = (__bf16)x1[2]; r[7] = (__bf16)x1[3];
    *(v8bf*)(dst + (size_t)i * 8) = r;
}

// ---------------------------------------------------------------------------
// Kernel 1: QKV projection on pre-converted bf16.  out = X @ W.T + bias.
// One 16x16 tile per wave, K=1024 over 32 bf16 WMMAs; inner loop is pure
// global_load_b128 + v_wmma (VALU-free steady state -> XDL-bound).
// Q,K written [B,NH,S,HD] bf16; V written transposed [B,NH,HD,S] bf16.
// ---------------------------------------------------------------------------
__global__ __launch_bounds__(256)
void proj_kernel(const __bf16* __restrict__ Xq, const __bf16* __restrict__ Xk,
                 const __bf16* __restrict__ Xv,
                 const __bf16* __restrict__ Wq, const float* __restrict__ bq,
                 const __bf16* __restrict__ Wk, const float* __restrict__ bk,
                 const __bf16* __restrict__ Wv, const float* __restrict__ bv,
                 __bf16* __restrict__ Qh, __bf16* __restrict__ Kh,
                 __bf16* __restrict__ Vt)
{
    const int lane = threadIdx.x & 31;
    const int wave = threadIdx.x >> 5;
    const int lo = lane & 15;
    const int hi = lane >> 4;

    const unsigned tilesPerMat = (BATCH * S_LEN / 16) * (HID / 16); // 512*64
    unsigned t = blockIdx.x * 8u + wave;                            // 0..98303
    int mat = (int)(t / tilesPerMat);
    unsigned r = t % tilesPerMat;
    int mt = (int)(r >> 6);
    int nt = (int)(r & 63);

    const __bf16* X  = (mat == 0) ? Xq : (mat == 1) ? Xk : Xv;
    const __bf16* W  = (mat == 0) ? Wq : (mat == 1) ? Wk : Wv;
    const float*  bi = (mat == 0) ? bq : (mat == 1) ? bk : bv;

    // A fragment row (per lane): X memory layout [S,B,H], row index b-major
    int m_row = mt * 16 + lo;
    int ab = m_row >> 10;
    int as = m_row & (S_LEN - 1);
    const __bf16* arow = X + ((size_t)as * BATCH + ab) * HID;
    // B fragment row (per lane): W is [N,K] row-major => computes A*W^T
    int n_row = nt * 16 + lo;
    const __bf16* brow = W + (size_t)n_row * HID;

    __builtin_prefetch(brow, 0, 2);   // global_prefetch_b8 on weight stream

    v8f acc = {0.f, 0.f, 0.f, 0.f, 0.f, 0.f, 0.f, 0.f};
    for (int kk = 0; kk < HID; kk += 32) {
        FragBF a, b;
        a.h[0] = *(const v8bf*)(arow + kk + 8 * hi);
        a.h[1] = *(const v8bf*)(arow + kk + 16 + 8 * hi);
        b.h[0] = *(const v8bf*)(brow + kk + 8 * hi);
        b.h[1] = *(const v8bf*)(brow + kk + 16 + 8 * hi);
        acc = wmma_bf16(a, b, acc);
    }

    int n_out = nt * 16 + lo;
    float bias = bi[n_out];
    int head = n_out >> 6;
    int d    = n_out & (HDIM - 1);
    for (int g = 0; g < 8; ++g) {
        int m_out = mt * 16 + g + 8 * hi;
        int ob = m_out >> 10;
        int os = m_out & (S_LEN - 1);
        __bf16 val = (__bf16)(acc[g] + bias);
        if (mat == 2) {
            Vt[((size_t)(ob * NHEAD + head) * HDIM + d) * S_LEN + os] = val;
        } else {
            __bf16* dst = (mat == 0) ? Qh : Kh;
            dst[((size_t)(ob * NHEAD + head) * S_LEN + os) * HDIM + d] = val;
        }
    }
}

// ---------------------------------------------------------------------------
// Kernel 2: attention. One workgroup per (b, head, 16-query-row tile).
// TDM stages the whole per-head K block (1024x64 bf16 = 128 KB) into LDS with
// one tensor_load_to_lds; phase-1 B fragments then come from ds_load_b128.
// ---------------------------------------------------------------------------
#define SC_STRIDE 1036                       // 1024 + 12 floats padding
#define KBUF_OFF  (16 * SC_STRIDE * 4)       // byte offset of kbuf in LDS

struct SMem {
    float  sc[16 * SC_STRIDE];               // 66304 B score buffer
    __bf16 kbuf[S_LEN * HDIM];               // 131072 B K head block
};                                           // 197376 B total (< 320 KB WGP)

__global__ __launch_bounds__(256)
void attn_kernel(const __bf16* __restrict__ Qh, const __bf16* __restrict__ Kh,
                 const __bf16* __restrict__ Vt,
                 const unsigned char* __restrict__ mask,
                 float* __restrict__ ctx_out, float* __restrict__ attw_out)
{
    __shared__ SMem smem;

    const int lane = threadIdx.x & 31;
    const int wave = threadIdx.x >> 5;
    const int lo = lane & 15;
    const int hi = lane >> 4;

    unsigned bid = blockIdx.x;               // B*NH*(S/16) = 8192 blocks
    int qt = (int)(bid & 63);
    int h  = (int)((bid >> 6) & (NHEAD - 1));
    int b  = (int)(bid >> 10);
    size_t headOff = (size_t)(b * NHEAD + h);

    // Q fragments (rows qt*16..+15, K-dim = HD = 64 -> two k32 fragments)
    const __bf16* qrow = Qh + (headOff * S_LEN + qt * 16 + lo) * HDIM;
    FragBF a0, a1;
    a0.h[0] = *(const v8bf*)(qrow + 8 * hi);
    a0.h[1] = *(const v8bf*)(qrow + 16 + 8 * hi);
    a1.h[0] = *(const v8bf*)(qrow + 32 + 8 * hi);
    a1.h[1] = *(const v8bf*)(qrow + 48 + 8 * hi);

    // ---- TDM: DMA per-head K block (128 KB, contiguous) into LDS -----------
#if __has_builtin(__builtin_amdgcn_tensor_load_to_lds)
    if (wave == 0) {
        unsigned long long ga =
            (unsigned long long)(const void*)(Kh + headOff * S_LEN * HDIM);
        // D# group 0: count=1 | lds_addr | global_addr[56:0] | type=2
        v4u g0;
        g0[0] = 1u;
        g0[1] = (unsigned)KBUF_OFF;
        g0[2] = (unsigned)(ga & 0xFFFFFFFFu);
        g0[3] = (unsigned)((ga >> 32) & 0x01FFFFFFu) | (2u << 30);
        // D# group 1: data_size=3 (8B); 1-D tile: tensor_dim0=tile_dim0=16384
        v8i g1;
        g1[0] = (int)(3u << 16);          // wg_mask=0 | data_size=3
        g1[1] = (int)(16384u << 16);      // tensor_dim0[15:0] @ bits[63:48]
        g1[2] = (int)(1u << 16);          // tensor_dim0 hi=0 | tensor_dim1=1
        g1[3] = (int)(16384u << 16);      // tensor_dim1 hi=0 | tile_dim0=16384
        g1[4] = 1;                        // tile_dim1=1, tile_dim2=0
        g1[5] = 16384;                    // tensor_dim0_stride lo
        g1[6] = 0;
        g1[7] = 0;
        v4i gz4 = {0, 0, 0, 0};
        v8i gz8 = {0, 0, 0, 0, 0, 0, 0, 0};
        __builtin_amdgcn_tensor_load_to_lds(g0, g1, gz4, gz4, gz8, 0);
        __builtin_amdgcn_s_wait_tensorcnt((short)0);
    }
#else
    // Fallback: cooperative copy, 16 B per thread per step
    {
        const v8bf* src = (const v8bf*)(Kh + headOff * S_LEN * HDIM);
        v8bf* dst = (v8bf*)smem.kbuf;
        for (int i = threadIdx.x; i < (S_LEN * HDIM) / 8; i += 256)
            dst[i] = src[i];
    }
#endif
    __syncthreads();

    // ---- Phase 1: scores -> LDS (K fragments from LDS via ds_load_b128) ----
    for (int jt = wave; jt < 64; jt += 8) {
        const __bf16* krow = smem.kbuf + (jt * 16 + lo) * HDIM;
        FragBF b0, b1;
        b0.h[0] = *(const v8bf*)(krow + 8 * hi);
        b0.h[1] = *(const v8bf*)(krow + 16 + 8 * hi);
        b1.h[0] = *(const v8bf*)(krow + 32 + 8 * hi);
        b1.h[1] = *(const v8bf*)(krow + 48 + 8 * hi);
        v8f acc = {0.f, 0.f, 0.f, 0.f, 0.f, 0.f, 0.f, 0.f};
        acc = wmma_bf16(a0, b0, acc);
        acc = wmma_bf16(a1, b1, acc);
        int j = jt * 16 + lo;
        float mv = mask[b * S_LEN + j] ? -1000000.0f : 0.0f;
        for (int g = 0; g < 8; ++g) {
            int m = g + 8 * hi;
            smem.sc[m * SC_STRIDE + jt * 16 + lo] = acc[g] * 0.125f + mv;
        }
    }
    __syncthreads();

    // ---- Phase 2: softmax per row, emit fp32 attn_weights -------------------
    {
        int rr = threadIdx.x >> 4;
        int ii = threadIdx.x & 15;
        float* row = smem.sc + rr * SC_STRIDE;
        float mx = -3.4e38f;
        for (int c = ii; c < S_LEN; c += 16) mx = fmaxf(mx, row[c]);
        for (int off = 8; off; off >>= 1) mx = fmaxf(mx, __shfl_xor(mx, off, 32));
        float sum = 0.f;
        for (int c = ii; c < S_LEN; c += 16) {
            float e = __expf(row[c] - mx);
            row[c] = e;
            sum += e;
        }
        for (int off = 8; off; off >>= 1) sum += __shfl_xor(sum, off, 32);
        float inv = 1.0f / sum;
        float* wout = attw_out + (headOff * S_LEN + (size_t)(qt * 16 + rr)) * S_LEN;
        for (int c = ii; c < S_LEN; c += 16) {
            float w = row[c] * inv;
            row[c] = w;
            wout[c] = w;
        }
    }
    __syncthreads();

    // ---- Phase 3: ctx = P(16x1024) @ V(1024x64), key-dim split across waves -
    int ct = wave & 3;
    int kh = wave >> 2;
    const __bf16* vrow = Vt + (headOff * HDIM + ct * 16 + lo) * S_LEN;
    v8f acc = {0.f, 0.f, 0.f, 0.f, 0.f, 0.f, 0.f, 0.f};
    for (int kk = kh * 512; kk < kh * 512 + 512; kk += 32) {
        FragBF a, bv;
        // on-the-fly fp32->bf16 A fragment from LDS scores
        {
            const float* p0 = smem.sc + lo * SC_STRIDE + kk + 8 * hi;
            const float* p1 = smem.sc + lo * SC_STRIDE + kk + 16 + 8 * hi;
            v4f x0 = *(const v4f*)(p0);
            v4f x1 = *(const v4f*)(p0 + 4);
            v4f x2 = *(const v4f*)(p1);
            v4f x3 = *(const v4f*)(p1 + 4);
            v8bf r0, r1;
            r0[0] = (__bf16)x0[0]; r0[1] = (__bf16)x0[1];
            r0[2] = (__bf16)x0[2]; r0[3] = (__bf16)x0[3];
            r0[4] = (__bf16)x1[0]; r0[5] = (__bf16)x1[1];
            r0[6] = (__bf16)x1[2]; r0[7] = (__bf16)x1[3];
            r1[0] = (__bf16)x2[0]; r1[1] = (__bf16)x2[1];
            r1[2] = (__bf16)x2[2]; r1[3] = (__bf16)x2[3];
            r1[4] = (__bf16)x3[0]; r1[5] = (__bf16)x3[1];
            r1[6] = (__bf16)x3[2]; r1[7] = (__bf16)x3[3];
            a.h[0] = r0;
            a.h[1] = r1;
        }
        bv.h[0] = *(const v8bf*)(vrow + kk + 8 * hi);
        bv.h[1] = *(const v8bf*)(vrow + kk + 16 + 8 * hi);
        acc = wmma_bf16(a, bv, acc);
    }
    __syncthreads();
    if (wave >= 4) {
        int base = ct * 256;
        for (int g = 0; g < 8; ++g)
            smem.sc[base + (g + 8 * hi) * 16 + lo] = acc[g];
    }
    __syncthreads();
    if (wave < 4) {
        int base = ct * 256;
        for (int g = 0; g < 8; ++g) {
            int m = g + 8 * hi;
            float v = acc[g] + smem.sc[base + m * 16 + lo];
            int s_row = qt * 16 + m;
            int d = ct * 16 + lo;
            ctx_out[((size_t)s_row * BATCH + b) * HID + h * HDIM + d] = v;
        }
    }
}

// ---------------------------------------------------------------------------
extern "C" void kernel_launch(void* const* d_in, const int* in_sizes, int n_in,
                              void* d_out, int out_size, void* d_ws, size_t ws_size,
                              hipStream_t stream)
{
    (void)in_sizes; (void)n_in; (void)out_size; (void)ws_size;

    const float* q_states = (const float*)d_in[0];
    const float* k_states = (const float*)d_in[1];
    const float* v_states = (const float*)d_in[2];
    const unsigned char* mask = (const unsigned char*)d_in[3];
    const float* Wq = (const float*)d_in[4];
    const float* bq = (const float*)d_in[5];
    const float* Wk = (const float*)d_in[6];
    const float* bk = (const float*)d_in[7];
    const float* Wv = (const float*)d_in[8];
    const float* bv = (const float*)d_in[9];

    float* ctx  = (float*)d_out;                               // [S,B,H]
    float* attw = ctx + (size_t)S_LEN * BATCH * HID;           // [B,NH,S,S]

    const size_t xElems = (size_t)S_LEN * BATCH * HID;         // 8388608
    const size_t wElems = (size_t)HID * HID;                   // 1048576
    const size_t headElems = (size_t)BATCH * NHEAD * S_LEN * HDIM;

    __bf16* Xqb = (__bf16*)d_ws;
    __bf16* Xkb = Xqb + xElems;
    __bf16* Xvb = Xkb + xElems;
    __bf16* Wqb = Xvb + xElems;
    __bf16* Wkb = Wqb + wElems;
    __bf16* Wvb = Wkb + wElems;
    __bf16* Qh  = Wvb + wElems;
    __bf16* Kh  = Qh + headElems;
    __bf16* Vt  = Kh + headElems;                              // ~102 MB total

    // fp32 -> bf16 conversion passes
    cvt_kernel<<<dim3((unsigned)(xElems / 8 / 256)), dim3(256), 0, stream>>>(q_states, Xqb, (int)(xElems / 8));
    cvt_kernel<<<dim3((unsigned)(xElems / 8 / 256)), dim3(256), 0, stream>>>(k_states, Xkb, (int)(xElems / 8));
    cvt_kernel<<<dim3((unsigned)(xElems / 8 / 256)), dim3(256), 0, stream>>>(v_states, Xvb, (int)(xElems / 8));
    cvt_kernel<<<dim3((unsigned)(wElems / 8 / 256)), dim3(256), 0, stream>>>(Wq, Wqb, (int)(wElems / 8));
    cvt_kernel<<<dim3((unsigned)(wElems / 8 / 256)), dim3(256), 0, stream>>>(Wk, Wkb, (int)(wElems / 8));
    cvt_kernel<<<dim3((unsigned)(wElems / 8 / 256)), dim3(256), 0, stream>>>(Wv, Wvb, (int)(wElems / 8));

    // 3 mats * 512 Mtiles * 64 Ntiles = 98304 wave-tiles / 8 per block
    proj_kernel<<<dim3(12288), dim3(256), 0, stream>>>(
        Xqb, Xkb, Xvb, Wqb, bq, Wkb, bk, Wvb, bv, Qh, Kh, Vt);

    // B * NH * (S/16) = 8192 blocks
    attn_kernel<<<dim3(8192), dim3(256), 0, stream>>>(
        Qh, Kh, Vt, mask, ctx, attw);
}
